// RNN_22402549416316
// MI455X (gfx1250) — matrix operand order repaired and checked by reading, not compiled, not run on hardware
//
#include <hip/hip_runtime.h>
#include <hip/hip_bf16.h>
#include <math.h>

// ---------------------------------------------------------------------------
// Types for CDNA5 WMMA (wave32): v_wmma_f32_16x16x32_bf16
// ---------------------------------------------------------------------------
typedef __attribute__((ext_vector_type(16))) __bf16 v16bf;
typedef __attribute__((ext_vector_type(8)))  __bf16 v8bf;
typedef __attribute__((ext_vector_type(8)))  float  v8f;
typedef __attribute__((ext_vector_type(4)))  int    v4i;

#define B_SZ   512
#define T_SZ   128
#define H_SZ   1024
#define NA     50
#define KX0    64      // layer-0 input width padded 50 -> 64 (multiple of 32)
#define UBF    0.3f

// Block tile for the recurrent-step GEMM
#define TM     128
#define TN     128
#define APITCH 40      // LDS row pitch in bf16 (32 + 8 pad, keeps 16B alignment)

#if __has_builtin(__builtin_amdgcn_global_load_async_to_lds_b128)
#define HAS_ASYNC_LDS 1
typedef __attribute__((address_space(1))) v4i g_v4i;   // global (device) int4
typedef __attribute__((address_space(3))) v4i l_v4i;   // LDS (shared) int4
#else
#define HAS_ASYNC_LDS 0
#endif

__device__ __forceinline__ void wait_async_zero() {
#if __has_builtin(__builtin_amdgcn_s_wait_asynccnt)
    __builtin_amdgcn_s_wait_asynccnt(0);
#else
    asm volatile("s_wait_asynccnt 0x0" ::: "memory");
#endif
}

__device__ __forceinline__ float fast_tanh(float x) {
#if __has_builtin(__builtin_amdgcn_tanhf)
    return __builtin_amdgcn_tanhf(x);
#else
    float r;
    asm volatile("v_tanh_f32 %0, %1" : "=v"(r) : "v"(x));
    return r;
#endif
}

__device__ __forceinline__ v16bf combine16(v8bf lo, v8bf hi) {
    v16bf r;
#pragma unroll
    for (int i = 0; i < 8; ++i) { r[i] = lo[i]; r[8 + i] = hi[i]; }
    return r;
}

// ---------------------------------------------------------------------------
// One recurrent step:  Hout = tanh( [Ax | Ah] * W^T + bias )
//   Ax: [512 x kx]   bf16 (row stride ax_stride)     -- input-projection part
//   Ah: [512 x 1024] bf16 (row stride ah_stride), may be null (t == 0)
//   W : [1024 x KW]  bf16 row-major, KW = kx + 1024  (concatenated W_ih | W_hh)
//   bias: [1024] f32 (b_ih + b_hh)
//
// 128x128 block tile staged through double-buffered LDS (async DMA to LDS on
// gfx1250); 8 waves arranged 4(M) x 2(N); each wave computes 32x64
// (2 M-tiles x 4 N-tiles, 8 f32 accumulators = 8 independent WMMA chains).
// Grid = (512/128, 1024/128) = (4, 8), block = 256 threads.
// ---------------------------------------------------------------------------
__global__ __launch_bounds__(256) void rnn_step_wmma(
    const __bf16* __restrict__ Ax, long ax_stride, int kx,
    const __bf16* __restrict__ Ah, long ah_stride,
    const __bf16* __restrict__ W, int KW,
    const float* __restrict__ bias,
    __bf16* __restrict__ Hout, long hout_stride)
{
    __shared__ __bf16 lsA[2][TM * APITCH];
    __shared__ __bf16 lsB[2][TN * APITCH];

    const int tid  = threadIdx.x;
    const int lane = tid & 31;
    const int wave = tid >> 5;
    const int lm   = lane & 15;     // tile column / A-row select
    const int lh   = lane >> 4;     // lane half
    const int wm   = wave >> 1;     // 0..3  (M position of wave in block)
    const int wn   = wave & 1;      // 0..1  (N position of wave in block)
    const int mblk = blockIdx.x * TM;
    const int nblk = blockIdx.y * TN;
    const int m0   = wm * 32;       // wave-local M offset
    const int n0   = wn * 64;       // wave-local N offset

    // Cooperative loader mapping: 2 threads per row, 16 bf16 (32B) each
    const int lrow  = tid >> 1;           // 0..127
    const int lhalf = (tid & 1) * 16;     // element offset within 32-wide k slab

    const __bf16* aSrcX = Ax + (long)(mblk + lrow) * ax_stride;
    // Fold -kx into the h base so that a single "+ kt*32" indexes either source.
    const __bf16* aSrcH = (Ah != nullptr) ? (Ah + (long)(mblk + lrow) * ah_stride - kx) : aSrcX;
    const __bf16* wSrc  = W + (long)(nblk + lrow) * KW;

    const int nkx = kx >> 5;
    const int nkh = (Ah != nullptr) ? (H_SZ >> 5) : 0;
    const int nkt = nkx + nkh;

    // Accumulators initialized with bias (C layout: n = lane%16, same across regs)
    v8f acc[2][4];
#pragma unroll
    for (int j = 0; j < 4; ++j) {
        const float bv = bias[nblk + n0 + j * 16 + lm];
#pragma unroll
        for (int i = 0; i < 8; ++i) { acc[0][j][i] = bv; acc[1][j][i] = bv; }
    }

    auto loadTile = [&](int kt, int buf) {
        // Branchless A-source select; W column offset is uniformly kt*32 since
        // the h-part base already absorbed -kx.
        const __bf16* base = (kt < nkx) ? aSrcX : aSrcH;
        const __bf16* asrc = base + (kt << 5) + lhalf;
        const __bf16* bsrc = wSrc + (kt << 5) + lhalf;
        __bf16* ad = &lsA[buf][lrow * APITCH + lhalf];
        __bf16* bd = &lsB[buf][lrow * APITCH + lhalf];
#if HAS_ASYNC_LDS
        g_v4i* ga = (g_v4i*)asrc;
        g_v4i* gb = (g_v4i*)bsrc;
        l_v4i* la = (l_v4i*)ad;
        l_v4i* lb = (l_v4i*)bd;
        __builtin_amdgcn_global_load_async_to_lds_b128(ga, la, 0,  0);
        __builtin_amdgcn_global_load_async_to_lds_b128(ga, la, 16, 0);
        __builtin_amdgcn_global_load_async_to_lds_b128(gb, lb, 0,  0);
        __builtin_amdgcn_global_load_async_to_lds_b128(gb, lb, 16, 0);
#else
        v8bf a0 = *(const v8bf*)(asrc);
        v8bf a1 = *(const v8bf*)(asrc + 8);
        v8bf b0 = *(const v8bf*)(bsrc);
        v8bf b1 = *(const v8bf*)(bsrc + 8);
        *(v8bf*)(ad)     = a0;
        *(v8bf*)(ad + 8) = a1;
        *(v8bf*)(bd)     = b0;
        *(v8bf*)(bd + 8) = b1;
#endif
    };

    // Per-lane LDS fragment offsets
    const int aoff0 = (m0 + lm) * APITCH + lh * 8;   // M-tile 0 (chunks at +0, +16)
    const int aoff1 = aoff0 + 16 * APITCH;           // M-tile 1
    const int boff  = (n0 + lm) * APITCH + lh * 8;   // N-tile 0 (chunks at +0, +16)

    loadTile(0, 0);
    for (int kt = 0; kt < nkt; ++kt) {
        const int buf = kt & 1;
#if HAS_ASYNC_LDS
        wait_async_zero();   // drain our async writes to `buf` before signaling
#endif
        __syncthreads();
        if (kt + 1 < nkt) loadTile(kt + 1, buf ^ 1);

        v16bf a0 = combine16(*(const v8bf*)&lsA[buf][aoff0],
                             *(const v8bf*)&lsA[buf][aoff0 + 16]);
        v16bf a1 = combine16(*(const v8bf*)&lsA[buf][aoff1],
                             *(const v8bf*)&lsA[buf][aoff1 + 16]);
#pragma unroll
        for (int j = 0; j < 4; ++j) {
            const int bo = boff + j * 16 * APITCH;
            v16bf bb = combine16(*(const v8bf*)&lsB[buf][bo],
                                 *(const v8bf*)&lsB[buf][bo + 16]);
            acc[0][j] = __builtin_amdgcn_wmma_f32_16x16x32_bf16(false, a0, false, bb, (short)0, acc[0][j], false, false);
            acc[1][j] = __builtin_amdgcn_wmma_f32_16x16x32_bf16(false, a1, false, bb, (short)0, acc[1][j], false, false);
        }
    }

    // Epilogue: fused tanh (native v_tanh_f32), store bf16 hidden state
#pragma unroll
    for (int mt = 0; mt < 2; ++mt) {
#pragma unroll
        for (int r = 0; r < 8; ++r) {
            const int gm = mblk + m0 + mt * 16 + lh * 8 + r;
            __bf16* orow = Hout + (long)gm * hout_stride;
#pragma unroll
            for (int j = 0; j < 4; ++j) {
                orow[nblk + n0 + j * 16 + lm] = (__bf16)fast_tanh(acc[mt][j][r]);
            }
        }
    }
}

// ---------------------------------------------------------------------------
// Prep kernels
// ---------------------------------------------------------------------------
__global__ void pad_x_kernel(const float* __restrict__ x, __bf16* __restrict__ xp) {
    // x: [512,128,50] f32  ->  xp: [512,128,64] bf16 (zero padded)
    long idx = (long)blockIdx.x * blockDim.x + threadIdx.x;
    if (idx >= (long)B_SZ * T_SZ * KX0) return;
    int k  = (int)(idx & (KX0 - 1));
    long bt = idx >> 6;
    xp[idx] = (k < NA) ? (__bf16)x[bt * NA + k] : (__bf16)0.0f;
}

__global__ void build_wb0_kernel(const float* __restrict__ wih0, const float* __restrict__ whh0,
                                 __bf16* __restrict__ wb) {
    // wb: [1024][64+1024]  row n = [w_ih0[n,0:50], 0..0, w_hh0[n,0:1024]]
    long idx = (long)blockIdx.x * blockDim.x + threadIdx.x;
    const int KW = KX0 + H_SZ;
    if (idx >= (long)H_SZ * KW) return;
    int c = (int)(idx % KW);
    int n = (int)(idx / KW);
    float v = (c < KX0) ? ((c < NA) ? wih0[n * NA + c] : 0.0f)
                        : whh0[n * H_SZ + (c - KX0)];
    wb[idx] = (__bf16)v;
}

__global__ void build_wb1_kernel(const float* __restrict__ wih1, const float* __restrict__ whh1,
                                 __bf16* __restrict__ wb) {
    // wb: [1024][2048]  row n = [w_ih1[n,:], w_hh1[n,:]]
    long idx = (long)blockIdx.x * blockDim.x + threadIdx.x;
    if (idx >= (long)H_SZ * 2 * H_SZ) return;
    int c = (int)(idx & (2 * H_SZ - 1));
    int n = (int)(idx >> 11);
    float v = (c < H_SZ) ? wih1[n * H_SZ + c] : whh1[n * H_SZ + (c - H_SZ)];
    wb[idx] = (__bf16)v;
}

__global__ void bias_sum_kernel(const float* a, const float* b, float* o, int n) {
    int i = blockIdx.x * blockDim.x + threadIdx.x;
    if (i < n) o[i] = a[i] + b[i];
}

// ---------------------------------------------------------------------------
// FC: logits[b,a] = h2_last[b,:] . fc_w[a,:] + fc_b[a]     (512 x 50, tiny)
// ---------------------------------------------------------------------------
__global__ void fc_kernel(const __bf16* __restrict__ h, const float* __restrict__ fc_w,
                          const float* __restrict__ fc_b, float* __restrict__ logits) {
    int idx = blockIdx.x * blockDim.x + threadIdx.x;
    if (idx >= B_SZ * NA) return;
    int b = idx / NA, a = idx % NA;
    const __bf16* hr = h + (long)b * H_SZ;
    const float*  wr = fc_w + (long)a * H_SZ;
    float s = fc_b[a];
    for (int k = 0; k < H_SZ; ++k) s += (float)hr[k] * wr[k];
    logits[idx] = s;
}

// ---------------------------------------------------------------------------
// Softmax + 16-iteration rebalance, one thread per sample (matches scan exactly)
// ---------------------------------------------------------------------------
__global__ void smax_rebal_kernel(const float* __restrict__ logits, float* __restrict__ out) {
    int b = blockIdx.x * blockDim.x + threadIdx.x;
    if (b >= B_SZ) return;
    const float* l = logits + (long)b * NA;
    float w[NA], old_[NA], wc[NA], wc2[NA];
    float mx = -3.402823e38f;
    for (int k = 0; k < NA; ++k) mx = fmaxf(mx, l[k]);
    float se = 0.f;
    for (int k = 0; k < NA; ++k) { w[k] = expf(l[k] - mx); se += w[k]; }
    for (int k = 0; k < NA; ++k) {
        w[k] /= se;
        old_[k] = w[k];
        wc[k]   = fminf(fmaxf(w[k], 0.0f), UBF);
    }
    for (int it = 0; it < 16; ++it) {
        float leftover = 0.f, snom = 0.f;
        for (int k = 0; k < NA; ++k) {
            leftover += old_[k] - wc[k];
            snom     += (wc[k] != UBF) ? wc[k] : 0.f;
        }
        bool cont = false;
        for (int k = 0; k < NA; ++k) {
            float nom = (wc[k] != UBF) ? wc[k] : 0.f;
            wc2[k] = wc[k] + leftover * nom / snom;
            if (wc2[k] > UBF) cont = true;
        }
        for (int k = 0; k < NA; ++k) {
            old_[k] = wc2[k];
            wc[k]   = cont ? fminf(fmaxf(wc2[k], 0.0f), UBF) : wc2[k];
        }
    }
    for (int k = 0; k < NA; ++k) out[(long)b * NA + k] = wc[k];
}

// ---------------------------------------------------------------------------
// Launch
// ---------------------------------------------------------------------------
extern "C" void kernel_launch(void* const* d_in, const int* in_sizes, int n_in,
                              void* d_out, int out_size, void* d_ws, size_t ws_size,
                              hipStream_t stream) {
    (void)in_sizes; (void)n_in; (void)out_size; (void)ws_size;
    const float* x     = (const float*)d_in[0];
    const float* w_ih0 = (const float*)d_in[1];
    const float* w_hh0 = (const float*)d_in[2];
    const float* b_ih0 = (const float*)d_in[3];
    const float* b_hh0 = (const float*)d_in[4];
    const float* w_ih1 = (const float*)d_in[5];
    const float* w_hh1 = (const float*)d_in[6];
    const float* b_ih1 = (const float*)d_in[7];
    const float* b_hh1 = (const float*)d_in[8];
    const float* fc_w  = (const float*)d_in[9];
    const float* fc_b  = (const float*)d_in[10];

    // Workspace carve-up (256B aligned)
    char* p = (char*)d_ws;
    auto carve = [&](size_t bytes) { char* r = p; p += (bytes + 255) & ~(size_t)255; return (void*)r; };
    __bf16* h1    = (__bf16*)carve((size_t)B_SZ * T_SZ * H_SZ * 2);       // layer-0 hidden seq
    __bf16* wb0   = (__bf16*)carve((size_t)H_SZ * (KX0 + H_SZ) * 2);      // [Wih0|Whh0] bf16
    __bf16* wb1   = (__bf16*)carve((size_t)H_SZ * (2 * H_SZ) * 2);        // [Wih1|Whh1] bf16
    __bf16* xp    = (__bf16*)carve((size_t)B_SZ * T_SZ * KX0 * 2);        // padded bf16 x
    float*  bias0 = (float*) carve((size_t)H_SZ * 4);
    float*  bias1 = (float*) carve((size_t)H_SZ * 4);
    __bf16* hA    = (__bf16*)carve((size_t)B_SZ * H_SZ * 2);              // layer-1 ping
    __bf16* hB    = (__bf16*)carve((size_t)B_SZ * H_SZ * 2);              // layer-1 pong
    float*  logits= (float*) carve((size_t)B_SZ * NA * 4);

    // --- prep ---
    {
        long n = (long)B_SZ * T_SZ * KX0;
        pad_x_kernel<<<(unsigned)((n + 255) / 256), 256, 0, stream>>>(x, xp);
        long n0 = (long)H_SZ * (KX0 + H_SZ);
        build_wb0_kernel<<<(unsigned)((n0 + 255) / 256), 256, 0, stream>>>(w_ih0, w_hh0, wb0);
        long n1 = (long)H_SZ * 2 * H_SZ;
        build_wb1_kernel<<<(unsigned)((n1 + 255) / 256), 256, 0, stream>>>(w_ih1, w_hh1, wb1);
        bias_sum_kernel<<<4, 256, 0, stream>>>(b_ih0, b_hh0, bias0, H_SZ);
        bias_sum_kernel<<<4, 256, 0, stream>>>(b_ih1, b_hh1, bias1, H_SZ);
    }

    const dim3 grid(B_SZ / TM, H_SZ / TN);  // (4, 8)
    const dim3 blk(256);
    const long TH = (long)T_SZ * H_SZ;

    // --- layer 0 recurrence: h1[:,t,:] = tanh([x_t | h1[:,t-1,:]] * WB0^T + bias0) ---
    for (int t = 0; t < T_SZ; ++t) {
        const __bf16* Ah = (t == 0) ? nullptr : (h1 + (long)(t - 1) * H_SZ);
        rnn_step_wmma<<<grid, blk, 0, stream>>>(
            xp + (long)t * KX0, (long)T_SZ * KX0, KX0,
            Ah, TH,
            wb0, KX0 + H_SZ, bias0,
            h1 + (long)t * H_SZ, TH);
    }

    // --- layer 1 recurrence: h2_t = tanh([h1[:,t,:] | h2_{t-1}] * WB1^T + bias1) ---
    for (int t = 0; t < T_SZ; ++t) {
        __bf16* cur        = (t & 1) ? hB : hA;
        const __bf16* prev = (t & 1) ? hA : hB;
        rnn_step_wmma<<<grid, blk, 0, stream>>>(
            h1 + (long)t * H_SZ, TH, H_SZ,
            (t == 0) ? nullptr : prev, (long)H_SZ,
            wb1, 2 * H_SZ, bias1,
            cur, (long)H_SZ);
    }
    // t = 127 (odd) -> final hidden state in hB

    fc_kernel<<<(B_SZ * NA + 255) / 256, 256, 0, stream>>>(hB, fc_w, fc_b, logits);
    smax_rebal_kernel<<<(B_SZ + 255) / 256, 256, 0, stream>>>(logits, (float*)d_out);
}
